// TopKSAE_30142080483458
// MI455X (gfx1250) — matrix operand order repaired and checked by reading, not compiled, not run on hardware
//
#include <hip/hip_runtime.h>
#include <hip/hip_bf16.h>

// ---------------------------------------------------------------------------
// TopK SAE forward for MI455X (gfx1250).
//   B=8192, D=1280, H=16384, K=64
//   d_out layout: [ rec (B*D) | hidden (B*H) | loss, recon_loss, sparsity, l0 ]
// Encoder pre-activations go into the hidden output region, then are replaced
// in-place by the sparse top-k ReLU result (no giant workspace needed).
// ---------------------------------------------------------------------------

#define SAE_B 8192
#define SAE_D 1280
#define SAE_H 16384

typedef __attribute__((ext_vector_type(16))) __bf16       v16bf;
typedef __attribute__((ext_vector_type(8)))  float        v8f;
typedef __attribute__((ext_vector_type(4)))  unsigned int v4u;
typedef __attribute__((ext_vector_type(8)))  unsigned int v8u;

// fp32 -> bf16 round-to-nearest-even, pack two into a dword
__device__ __forceinline__ unsigned int pack_bf16(float a, float b) {
    unsigned int ua = __float_as_uint(a);
    unsigned int ub = __float_as_uint(b);
    ua += 0x7FFFu + ((ua >> 16) & 1u);
    ub += 0x7FFFu + ((ub >> 16) & 1u);
    return (ua >> 16) | (ub & 0xFFFF0000u);
}

__device__ __forceinline__ v16bf make_frag(const unsigned int* p0,
                                           const unsigned int* p1) {
    v4u lo = *(const v4u*)p0;
    v4u hi = *(const v4u*)p1;
    v8u w;
#pragma unroll
    for (int j = 0; j < 4; ++j) { w[j] = lo[j]; w[4 + j] = hi[j]; }
    return __builtin_bit_cast(v16bf, w);
}

// Load one 32-float row slab (128B, two cachelines), optional per-K bias
// subtraction, pack to 16 bf16-pair dwords held in registers.
__device__ __forceinline__ void load_slab32(const float* __restrict__ p,
                                            const float* __restrict__ subp,
                                            v4u out[4]) {
    float4 f[8];
#pragma unroll
    for (int j = 0; j < 8; ++j) f[j] = ((const float4*)p)[j];
    if (subp) {
        const float4* sp = (const float4*)subp;
#pragma unroll
        for (int j = 0; j < 8; ++j) {
            float4 s = sp[j];
            f[j].x -= s.x; f[j].y -= s.y; f[j].z -= s.z; f[j].w -= s.w;
        }
    }
#pragma unroll
    for (int j = 0; j < 4; ++j) {
        out[j][0] = pack_bf16(f[2 * j].x,     f[2 * j].y);
        out[j][1] = pack_bf16(f[2 * j].z,     f[2 * j].w);
        out[j][2] = pack_bf16(f[2 * j + 1].x, f[2 * j + 1].y);
        out[j][3] = pack_bf16(f[2 * j + 1].z, f[2 * j + 1].w);
    }
}

// ---------------------------------------------------------------------------
// NT GEMM:  C[M,N] = A[M,K] * Bm[N,K]^T (+ bias1[N] + bias2[N]),
// optionally A' = A - sub[K] (broadcast along rows).
// Block tile: 128 x BN x 32, 256 threads = 8 waves (2 M-bands x 4 N-bands).
// Each wave owns a 64 x (BN/4) patch = 4 x (BN/64) grid of 16x16 WMMA tiles,
// v_wmma_f32_16x16x32_bf16 with f32 accumulate.
// Software pipelined: next K-slab is global-loaded + bf16-packed into
// registers while the current LDS tile feeds the WMMAs.
// ---------------------------------------------------------------------------
template <int BN>
__global__ __launch_bounds__(256) void sae_gemm_nt_bf16(
    const float* __restrict__ A, const float* __restrict__ Bm,
    const float* __restrict__ sub, const float* __restrict__ bias1,
    const float* __restrict__ bias2, float* __restrict__ C,
    int M, int N, int Kd)
{
    constexpr int NSUB  = BN / 64;     // 16-wide N-subtiles per wave
    constexpr int SLOTS = 128 + BN;    // staged rows per K-step
    // Row stride 20 dwords (80B): 16B-aligned for b128 ds ops, breaks the
    // worst bank alignment.
    __shared__ __align__(16) unsigned int As[128 * 20];
    __shared__ __align__(16) unsigned int Bs[BN * 20];

    const int tid   = threadIdx.x;
    const int nTile = blockIdx.x * BN;
    const int mTile = blockIdx.y * 128;

    const int lane = tid & 31;
    const int wave = tid >> 5;
    const int wm   = wave >> 2;        // 0..1 : 64-row band
    const int wn   = wave & 3;         // 0..3 : (BN/4)-col band
    const int lr   = lane & 15;
    const int hk   = lane >> 4;        // lane half selects K interleave

    v8f acc[4][NSUB];
#pragma unroll
    for (int m = 0; m < 4; ++m)
#pragma unroll
        for (int n = 0; n < NSUB; ++n)
#pragma unroll
            for (int j = 0; j < 8; ++j) acc[m][n][j] = 0.0f;

    // Staging: slot r < 128 -> A row r; slot r >= 128 -> B row (r-128).
    const int slot0 = tid;
    const int slot1 = tid + 256;       // only valid when SLOTS > 256

    auto slab_src = [&](int r, int k0) -> const float* {
        return (r < 128) ? A + (size_t)(mTile + r) * Kd + k0
                         : Bm + (size_t)(nTile + (r - 128)) * Kd + k0;
    };

    v4u reg0[4], reg1[4];
    load_slab32(slab_src(slot0, 0), (slot0 < 128 && sub) ? sub : nullptr, reg0);
    if (SLOTS > 256 && slot1 < SLOTS)
        load_slab32(slab_src(slot1, 0), nullptr, reg1);

    for (int k0 = 0; k0 < Kd; k0 += 32) {
        __syncthreads();   // previous compute finished reading LDS
        {
            unsigned int* d0 = (slot0 < 128) ? &As[slot0 * 20]
                                             : &Bs[(slot0 - 128) * 20];
#pragma unroll
            for (int j = 0; j < 4; ++j) *(v4u*)&d0[j * 4] = reg0[j];
            if (SLOTS > 256 && slot1 < SLOTS) {
                unsigned int* d1 = &Bs[(slot1 - 128) * 20];
#pragma unroll
                for (int j = 0; j < 4; ++j) *(v4u*)&d1[j * 4] = reg1[j];
            }
        }
        __syncthreads();

        // kick off next slab's global loads; they complete under the WMMAs
        if (k0 + 32 < Kd) {
            load_slab32(slab_src(slot0, k0 + 32),
                        (slot0 < 128 && sub) ? sub + k0 + 32 : nullptr, reg0);
            if (SLOTS > 256 && slot1 < SLOTS)
                load_slab32(slab_src(slot1, k0 + 32), nullptr, reg1);
        }

        // B fragments once per K-step; A fragment per m to cap VGPR use
        v16bf bF[NSUB];
#pragma unroll
        for (int n = 0; n < NSUB; ++n) {
            const int Nr = wn * (NSUB * 16) + n * 16 + lr;
            bF[n] = make_frag(&Bs[Nr * 20 + hk * 4], &Bs[Nr * 20 + 8 + hk * 4]);
        }
#pragma unroll
        for (int m = 0; m < 4; ++m) {
            const int Mr = wm * 64 + m * 16 + lr;
            v16bf aF = make_frag(&As[Mr * 20 + hk * 4],
                                 &As[Mr * 20 + 8 + hk * 4]);
#pragma unroll
            for (int n = 0; n < NSUB; ++n)
                acc[m][n] = __builtin_amdgcn_wmma_f32_16x16x32_bf16(
                    false, aF, false, bF[n], (short)0, acc[m][n],
                    false, false);
        }
    }

    // ---- epilogue: bias + store (16-lane contiguous column groups) ----
#pragma unroll
    for (int m = 0; m < 4; ++m)
#pragma unroll
        for (int n = 0; n < NSUB; ++n) {
            const int col  = nTile + wn * (NSUB * 16) + n * 16 + lr;
            const int rowb = mTile + wm * 64 + m * 16 + hk * 8;
            float bv = 0.0f;
            if (bias1) bv += bias1[col];
            if (bias2) bv += bias2[col];
#pragma unroll
            for (int r = 0; r < 8; ++r)
                C[(size_t)(rowb + r) * (size_t)N + col] = acc[m][n][r] + bv;
        }
}

// ---------------------------------------------------------------------------
// Exact per-row top-k via 4-pass radix select on order-preserving uint keys.
// One workgroup per row; row (64KB) lives in dynamic LDS. In-place rewrite:
// pre -> sparse relu(hidden). Also accumulates l0 (mean active count).
// ---------------------------------------------------------------------------
__global__ __launch_bounds__(256) void sae_topk_inplace(
    float* __restrict__ hid, const int* __restrict__ kptr,
    float* __restrict__ l0_out)
{
    extern __shared__ unsigned int smem[];
    unsigned int* keys = smem;                 // [16384]
    unsigned int* hist = smem + SAE_H;         // [256]
    unsigned int* ctl  = smem + SAE_H + 256;   // prefix, remain, eq-claim
    float*        scnt = (float*)(smem + SAE_H + 259);

    const int tid = threadIdx.x;
    float* rowp = hid + (size_t)blockIdx.x * SAE_H;

    for (int i = tid; i < SAE_H; i += 256) {
        unsigned int b = __float_as_uint(rowp[i]);
        keys[i] = (b & 0x80000000u) ? ~b : (b | 0x80000000u);
    }
    if (tid == 0) { ctl[0] = 0u; ctl[1] = (unsigned int)(*kptr); ctl[2] = 0u; *scnt = 0.0f; }
    __syncthreads();

    for (int shift = 24; shift >= 0; shift -= 8) {
        hist[tid] = 0u;
        __syncthreads();
        const unsigned int pfx    = ctl[0];
        const unsigned int hiMask = (shift == 24) ? 0u : (0xFFFFFFFFu << (shift + 8));
        for (int i = tid; i < SAE_H; i += 256) {
            unsigned int u = keys[i];
            if ((u & hiMask) == pfx) atomicAdd(&hist[(u >> shift) & 0xFFu], 1u);
        }
        __syncthreads();
        if (tid == 0) {
            unsigned int need = ctl[1], cum = 0;
            int b = 255;
            for (;; --b) {
                unsigned int c = hist[b];
                if (cum + c >= need || b == 0) break;
                cum += c;
            }
            ctl[0] = pfx | ((unsigned int)b << shift);
            ctl[1] = need - cum;
        }
        __syncthreads();
    }

    const unsigned int T = ctl[0];       // exact k-th largest key
    const unsigned int needEq = ctl[1];  // how many at-threshold to keep
    float cnt = 0.0f;
    for (int i = tid; i < SAE_H; i += 256) {
        unsigned int u = keys[i];
        bool sel = false;
        if (u > T) sel = true;
        else if (u == T) sel = (atomicAdd(&ctl[2], 1u) < needEq);
        unsigned int fb = (u & 0x80000000u) ? (u ^ 0x80000000u) : ~u;
        float f = __uint_as_float(fb);
        float o = sel ? fmaxf(f, 0.0f) : 0.0f;
        rowp[i] = o;
        if (o > 0.0f) cnt += 1.0f;
    }
    atomicAdd(scnt, cnt);
    __syncthreads();
    if (tid == 0) atomicAdd(l0_out, (*scnt) * (1.0f / (float)SAE_B));
}

// ---------------------------------------------------------------------------
__global__ void sae_init_scalars(float* s) {
    if (threadIdx.x < 4) s[threadIdx.x] = 0.0f;
}

__global__ __launch_bounds__(256) void sae_mse(
    const float* __restrict__ rec, const float* __restrict__ x,
    float* __restrict__ loss, float* __restrict__ rloss, long long n)
{
    __shared__ float red[256];
    float acc = 0.0f;
    for (long long i = (long long)blockIdx.x * blockDim.x + threadIdx.x; i < n;
         i += (long long)gridDim.x * blockDim.x) {
        float d = rec[i] - x[i];
        acc += d * d;
    }
    red[threadIdx.x] = acc;
    __syncthreads();
    for (int s = 128; s > 0; s >>= 1) {
        if (threadIdx.x < s) red[threadIdx.x] += red[threadIdx.x + s];
        __syncthreads();
    }
    if (threadIdx.x == 0) {
        float v = red[0] * (1.0f / ((float)SAE_B * (float)SAE_D));
        atomicAdd(loss, v);
        atomicAdd(rloss, v);
    }
}

// ---------------------------------------------------------------------------
extern "C" void kernel_launch(void* const* d_in, const int* in_sizes, int n_in,
                              void* d_out, int out_size, void* d_ws, size_t ws_size,
                              hipStream_t stream) {
    (void)in_sizes; (void)n_in; (void)out_size; (void)d_ws; (void)ws_size;
    const float* x     = (const float*)d_in[0];
    const float* W_enc = (const float*)d_in[1];
    const float* b_enc = (const float*)d_in[2];
    const float* W_dec = (const float*)d_in[3];
    const float* b_dec = (const float*)d_in[4];
    const float* b_pre = (const float*)d_in[5];
    const int*   kptr  = (const int*)d_in[6];

    float* out  = (float*)d_out;
    float* rec  = out;                                        // [B,D]
    float* hid  = out + (size_t)SAE_B * SAE_D;                // [B,H]
    float* scal = hid + (size_t)SAE_B * SAE_H;                // 4 scalars

    sae_init_scalars<<<1, 32, 0, stream>>>(scal);

    // encode: pre = (x - b_pre) @ W_enc^T + b_enc  -> hidden region
    sae_gemm_nt_bf16<256><<<dim3(SAE_H / 256, SAE_B / 128), 256, 0, stream>>>(
        x, W_enc, b_pre, b_enc, nullptr, hid, SAE_B, SAE_H, SAE_D);

    // exact top-k + relu, in-place; accumulates l0 into scal[3]
    const size_t tk_lds = (size_t)(SAE_H + 256 + 4) * sizeof(unsigned int);
    sae_topk_inplace<<<SAE_B, 256, tk_lds, stream>>>(hid, kptr, &scal[3]);

    // decode: rec = hidden @ W_dec^T + b_dec + b_pre (BN=128 -> 640 WGs)
    sae_gemm_nt_bf16<128><<<dim3(SAE_D / 128, SAE_B / 128), 256, 0, stream>>>(
        hid, W_dec, nullptr, b_dec, b_pre, rec, SAE_B, SAE_D, SAE_H);

    // losses: loss == reconstruction_loss; sparsity stays 0 from init
    sae_mse<<<2048, 256, 0, stream>>>(rec, x, &scal[0], &scal[1],
                                      (long long)SAE_B * SAE_D);
}
